// GATE_46626164965827
// MI455X (gfx1250) — matrix operand order repaired and checked by reading, not compiled, not run on hardware
//
#include <hip/hip_runtime.h>

// ---------------- problem constants (match reference) ----------------
#define N_NODES 50000
#define N_EDGES 800000
#define IN_DIM  512
#define HID_DIM 128
#define OUT_DIM 32

typedef __attribute__((ext_vector_type(16))) __bf16 v16bf;
typedef __attribute__((ext_vector_type(4)))  __bf16 v4bf;
typedef __attribute__((ext_vector_type(8)))  float  v8f;

__device__ __forceinline__ float elu1(float x) {
    return x > 0.f ? x : (__expf(x) - 1.f);
}

// ---------------------------------------------------------------------
// WMMA GEMM: C(MxN) = act(A)(MxK) * B_eff(KxN), fp32 in/out, bf16 compute,
// fp32 accumulate (v_wmma_f32_16x16x32_bf16). Compile-time shapes; LDS
// double-buffered (global fetch of K-tile t+1 overlaps WMMA on tile t);
// tiles staged in LDS in WMMA operand order -> operand fetch is one
// contiguous 32B LDS read (2x ds_load_b128); global staging is float4
// vectorized (b128 loads).
//
// Operand-order maps (wave32, v_wmma_f32_16x16x32_bf16):
//   A 16x32:  lane l, elem i -> m = l&15, k = ((l>>4)<<4) + i  (same k order as B)
//   B 32x16:  lane l, elem i -> n = l&15, k = ((l>>4)<<4) + i
//   C/D:      vgpr i, lane l -> m = (l<16 ? i : i+8), n = l&15
//
// block = (32, WPB): each wave owns one 16x16 tile;
// blockIdx.x = row tile, blockIdx.y*WPB + wave = col tile. M%16==0, K%32==0.
// ---------------------------------------------------------------------
template <int WPB, int K, int N, bool TRANSB, bool ELUA>
__global__ void gat_gemm_wmma_bf16(const float* __restrict__ A,
                                   const float* __restrict__ B,
                                   float* __restrict__ C)
{
    constexpr int NT = 32 * WPB;          // threads per block
    constexpr int BE = WPB * 512;         // B-tile elements per K-step
    constexpr int QB = BE / 4;            // B-tile quads (QB/NT == 4 exactly)
    constexpr int KT = K / 32;            // K-steps

    __shared__ alignas(32) __bf16 ldsA[2][512];
    __shared__ alignas(32) __bf16 ldsB[2][BE];

    const int lane    = threadIdx.x;                  // 0..31
    const int wave    = threadIdx.y;                  // 0..WPB-1
    const int tid     = wave * 32 + lane;
    const int rowBase = blockIdx.x * 16;
    const int colBase = blockIdx.y * (WPB << 4);

    // A tile: 128 quads; quad q -> dest elems j=4q..4q+3, one float4 load.
    auto stageA = [&](int k0, __bf16* dst) {
#pragma unroll
        for (int q0 = 0; q0 < 128; q0 += NT) {
            int q = q0 + tid;
            if (NT > 128 && q >= 128) break;          // NT=256: waves 4..7 idle
            int l  = q >> 2;                          // 0..31
            int i4 = q & 3;
            int m  = l & 15;
            int kb = ((l >> 4) << 4) + (i4 << 2);
            float4 v = *(const float4*)(&A[(size_t)(rowBase + m) * K + (k0 + kb)]);
            if (ELUA) { v.x = elu1(v.x); v.y = elu1(v.y); v.z = elu1(v.z); v.w = elu1(v.w); }
            v4bf o = { (__bf16)v.x, (__bf16)v.y, (__bf16)v.z, (__bf16)v.w };
            *(v4bf*)(dst + (q << 2)) = o;
        }
    };

    auto stageB = [&](int k0, __bf16* dst) {
        if (TRANSB) {
            // elem index ~ consecutive k, row-major stride K: full vector path
#pragma unroll
            for (int q0 = 0; q0 < QB; q0 += NT) {
                int q  = q0 + tid;
                int wv = q >> 7;
                int l  = (q >> 2) & 31;
                int i4 = q & 3;
                int n  = colBase + (wv << 4) + (l & 15);
                int k  = k0 + ((l >> 4) << 4) + (i4 << 2);
                float4 v = *(const float4*)(&B[(size_t)n * K + k]);
                v4bf o = { (__bf16)v.x, (__bf16)v.y, (__bf16)v.z, (__bf16)v.w };
                *(v4bf*)(dst + (q << 2)) = o;
            }
        } else {
            // vectorize along contiguous n; scatter 4 b16 into LDS
#pragma unroll
            for (int q0 = 0; q0 < QB; q0 += NT) {
                int q    = q0 + tid;
                int wv   = q >> 7;
                int rem  = q & 127;
                int kidx = rem >> 2;                  // 0..31
                int n4   = rem & 3;
                int nb   = colBase + (wv << 4) + (n4 << 2);
                float4 v = *(const float4*)(&B[(size_t)(k0 + kidx) * N + nb]);
                float vv[4] = { v.x, v.y, v.z, v.w };
                int lhalf = (kidx >> 4) << 4;
                int i     = kidx & 15;
#pragma unroll
                for (int t = 0; t < 4; ++t) {
                    int ld = lhalf + (n4 << 2) + t;   // dest lane
                    dst[(wv << 9) + (ld << 4) + i] = (__bf16)vv[t];
                }
            }
        }
    };

    stageA(0, ldsA[0]);
    stageB(0, ldsB[0]);
    __syncthreads();

    v8f acc = {};
#pragma unroll
    for (int kt = 0; kt < KT; ++kt) {
        const int cur = kt & 1;
        // overlap: fetch next K-tile into the other buffer
        if (kt + 1 < KT) {
            stageA((kt + 1) * 32, ldsA[cur ^ 1]);
            stageB((kt + 1) * 32, ldsB[cur ^ 1]);
        }
        // contiguous 32B operand fetches (2x ds_load_b128 each)
        v16bf av = *(const v16bf*)(&ldsA[cur][lane << 4]);
        v16bf bv = *(const v16bf*)(&ldsB[cur][(wave << 9) + (lane << 4)]);
        acc = __builtin_amdgcn_wmma_f32_16x16x32_bf16(
                  /*neg_a=*/false, av, /*neg_b=*/false, bv,
                  /*c_mod=*/(short)0, acc, /*reuse_a=*/false, /*reuse_b=*/false);
        __syncthreads();   // next-tile staging done; cur buffer fully consumed
    }

    // ---- store D (two coalesced 64B runs per VGPR row) ----
    int n     = colBase + (wave << 4) + (lane & 15);
    int mrow0 = (lane < 16) ? 0 : 8;
#pragma unroll
    for (int i = 0; i < 8; ++i)
        C[(size_t)(rowBase + mrow0 + i) * N + n] = acc[i];
}

// ---------------- per-node attention logits: warp per node ----------------
__global__ void gat_node_logits(const float* __restrict__ h1p,
                                const float* __restrict__ att_src,
                                const float* __restrict__ att_dst,
                                float* __restrict__ a_src,
                                float* __restrict__ a_dst)
{
    int gid  = blockIdx.x * blockDim.x + threadIdx.x;
    int node = gid >> 5;
    int lane = gid & 31;
    if (node >= N_NODES) return;
    const float* row = h1p + (size_t)node * HID_DIM;
    float s0 = 0.f, s1 = 0.f;
#pragma unroll
    for (int c = 0; c < HID_DIM; c += 32) {
        float v = row[c + lane];
        s0 += v * att_src[c + lane];
        s1 += v * att_dst[c + lane];
    }
#pragma unroll
    for (int off = 16; off; off >>= 1) {
        s0 += __shfl_xor(s0, off, 32);
        s1 += __shfl_xor(s1, off, 32);
    }
    if (lane == 0) { a_src[node] = s0; a_dst[node] = s1; }
}

// ---------------- zero-fill ----------------
__global__ void gat_zero(float* __restrict__ p, int n)
{
    int i = blockIdx.x * blockDim.x + threadIdx.x;
    if (i < n) p[i] = 0.f;
}

// ---------------- edge pass 1: sigmoid logit + segment max ----------------
// sigmoid > 0, so uint-bit atomicMax == float max (m initialized to 0 bits).
__global__ void gat_edge_logit_max(const int* __restrict__ src,
                                   const int* __restrict__ dst,
                                   const float* __restrict__ a_src,
                                   const float* __restrict__ a_dst,
                                   float* __restrict__ ew,
                                   unsigned int* __restrict__ mbits)
{
    int e = blockIdx.x * blockDim.x + threadIdx.x;
    if (e >= N_EDGES) return;
    float x  = a_src[src[e]] + a_dst[dst[e]];
    float lg = 1.f / (1.f + __expf(-x));
    ew[e] = lg;
    atomicMax(&mbits[dst[e]], __float_as_uint(lg));
}

// ---------------- edge pass 2: exp(e - m[dst]) + segment sum ----------------
__global__ void gat_edge_exp_sum(const int* __restrict__ dst,
                                 const float* __restrict__ m,
                                 float* __restrict__ ew,
                                 float* __restrict__ ssum)
{
    int e = blockIdx.x * blockDim.x + threadIdx.x;
    if (e >= N_EDGES) return;
    float ex = __expf(ew[e] - m[dst[e]]);
    ew[e] = ex;
    atomicAdd(&ssum[dst[e]], ex);
}

// ---------------- edge aggregation: warp per edge, 128-dim rows ----------------
__global__ void gat_edge_aggregate(const int* __restrict__ src,
                                   const int* __restrict__ dst,
                                   const float* __restrict__ ew,
                                   const float* __restrict__ ssum,
                                   const float* __restrict__ x,
                                   float* __restrict__ out)
{
    int gid  = blockIdx.x * blockDim.x + threadIdx.x;
    int e    = gid >> 5;
    int lane = gid & 31;
    if (e >= N_EDGES) return;
    int si = src[e], di = dst[e];
    float w = ew[e] / (ssum[di] + 1e-16f);
    const float* xr   = x   + (size_t)si * HID_DIM;
    float*       orow = out + (size_t)di * HID_DIM;
#pragma unroll
    for (int c = 0; c < HID_DIM; c += 32)
        atomicAdd(&orow[c + lane], xr[c + lane] * w);
}

// ---------------------------------------------------------------------
extern "C" void kernel_launch(void* const* d_in, const int* in_sizes, int n_in,
                              void* d_out, int out_size, void* d_ws, size_t ws_size,
                              hipStream_t stream)
{
    (void)in_sizes; (void)n_in; (void)out_size; (void)ws_size;

    const float* features = (const float*)d_in[0];   // N x 512
    const float* W1       = (const float*)d_in[1];   // 512 x 128
    const float* W2       = (const float*)d_in[2];   // 128 x 32
    const float* att_src  = (const float*)d_in[3];   // 128
    const float* att_dst  = (const float*)d_in[4];   // 128
    const int*   ei       = (const int*)d_in[5];     // 2 x E
    const int*   e_src    = ei;
    const int*   e_dst    = ei + N_EDGES;

    float* outp = (float*)d_out;
    float* h2   = outp;                               // N x 32
    float* h4   = outp + (size_t)N_NODES * OUT_DIM;   // N x 512

    // workspace layout
    float* ws    = (float*)d_ws;
    float* h1p   = ws;                                   // N x 128 (reused for h3p)
    float* agg   = h1p + (size_t)N_NODES * HID_DIM;      // N x 128
    float* a_src = agg + (size_t)N_NODES * HID_DIM;      // N
    float* a_dst = a_src + N_NODES;                      // N
    float* mbuf  = a_dst + N_NODES;                      // N (max, bits-init 0)
    float* sbuf  = mbuf + N_NODES;                       // N (sum)
    float* ew    = sbuf + N_NODES;                       // E (logit -> exp, reused)

    const int ROWT = N_NODES / 16;                       // 3125 row tiles
    dim3 blk8(32, 8), blk2(32, 2);

    // 1) h1p = X @ W1            (50000x512x128)
    gat_gemm_wmma_bf16<8, IN_DIM, HID_DIM, false, false>
        <<<dim3(ROWT, 1), blk8, 0, stream>>>(features, W1, h1p);

    // 2) per-node attention logits
    gat_node_logits<<<(N_NODES * 32 + 255) / 256, 256, 0, stream>>>(
        h1p, att_src, att_dst, a_src, a_dst);

    // 3) init max/sum (2N contiguous) and agg accumulator
    gat_zero<<<(2 * N_NODES + 255) / 256, 256, 0, stream>>>(mbuf, 2 * N_NODES);
    gat_zero<<<(N_NODES * HID_DIM + 255) / 256, 256, 0, stream>>>(agg, N_NODES * HID_DIM);

    // 4) edge softmax (weights shared by conv1 and conv3)
    gat_edge_logit_max<<<(N_EDGES + 255) / 256, 256, 0, stream>>>(
        e_src, e_dst, a_src, a_dst, ew, (unsigned int*)mbuf);
    gat_edge_exp_sum<<<(N_EDGES + 255) / 256, 256, 0, stream>>>(
        e_dst, mbuf, ew, sbuf);

    // 5) agg = segment_sum(h1p[src] * a, dst)
    gat_edge_aggregate<<<(N_EDGES * 32 + 255) / 256, 256, 0, stream>>>(
        e_src, e_dst, ew, sbuf, h1p, agg);

    // 6) h2 = elu(agg) @ W2      (50000x128x32) -> output
    gat_gemm_wmma_bf16<2, HID_DIM, OUT_DIM, false, true>
        <<<dim3(ROWT, 1), blk2, 0, stream>>>(agg, W2, h2);

    // 7) h3p = h2 @ W2^T         (50000x32x128) -> reuse h1p buffer
    gat_gemm_wmma_bf16<8, OUT_DIM, HID_DIM, true, false>
        <<<dim3(ROWT, 1), blk8, 0, stream>>>(h2, W2, h1p);

    // 8) second propagation with the SAME edge weights
    gat_zero<<<(N_NODES * HID_DIM + 255) / 256, 256, 0, stream>>>(agg, N_NODES * HID_DIM);
    gat_edge_aggregate<<<(N_EDGES * 32 + 255) / 256, 256, 0, stream>>>(
        e_src, e_dst, ew, sbuf, h1p, agg);

    // 9) h4 = elu(agg) @ W1^T    (50000x128x512) -> output
    gat_gemm_wmma_bf16<8, HID_DIM, IN_DIM, true, true>
        <<<dim3(ROWT, 4), blk8, 0, stream>>>(agg, W1, h4);
}
